// SelfAttention_18674517803700
// MI455X (gfx1250) — matrix operand order repaired and checked by reading, not compiled, not run on hardware
//
#include <hip/hip_runtime.h>
#include <hip/hip_bf16.h>

#define DEV __device__ __forceinline__
#define AS1 __attribute__((address_space(1)))
#define AS3 __attribute__((address_space(3)))

typedef __attribute__((ext_vector_type(16))) _Float16 v16h;
typedef __attribute__((ext_vector_type(8)))  _Float16 v8h;
typedef __attribute__((ext_vector_type(8)))  float    v8f;
typedef __attribute__((ext_vector_type(4)))  int      v4i;

namespace cfg {
constexpr int B  = 2;
constexpr int T  = 2048;
constexpr int D  = 2048;
constexpr int NH = 16;
constexpr int HD = 128;   // head dim
constexpr int FR = 64;    // rotary freqs
constexpr int TD = 3 * D; // 6144
}

union FragH { v16h v; _Float16 h[16]; };
union FragF { v8f  v; float    f[8];  };

// ---------------------------------------------------------------------------
// Async global->LDS (CDNA5 GLOBAL_LOAD_ASYNC_TO_LDS_B128, ASYNCcnt-tracked).
// Builtin signature (probed via hipcc diagnostic): (v4i AS1*, v4i AS3*, imm, imm)
// ---------------------------------------------------------------------------
#if defined(__gfx1250__) && \
    __has_builtin(__builtin_amdgcn_global_load_async_to_lds_b128) && \
    __has_builtin(__builtin_amdgcn_s_wait_asynccnt)
#define HAS_ASYNC_LDS 1
#else
#define HAS_ASYNC_LDS 0
#endif

DEV void async_copy16(const _Float16* gsrc, _Float16* ldst) {
#if HAS_ASYNC_LDS
  __builtin_amdgcn_global_load_async_to_lds_b128(
      (AS1 v4i*)(_Float16*)gsrc, (AS3 v4i*)ldst, 0, 0);
#else
  *(v8h*)ldst = *(const v8h*)gsrc;
#endif
}

DEV void async_join() {
#if HAS_ASYNC_LDS
  __builtin_amdgcn_s_wait_asynccnt(0);
#endif
}

// A fragment (16x32 f16), LDS row-major [row][k], lane half g holds k = 16g..16g+15 (contiguous).
DEV v16h frag_a_lds(const _Float16* base, int row0, int ldm, int lane) {
  const int m = lane & 15, g = lane >> 4;
  const _Float16* p = base + (size_t)(row0 + m) * ldm;
  FragH fr;
#pragma unroll
  for (int v = 0; v < 8; ++v) {
    const int k = 2 * v + 16 * g;
    fr.h[2 * v]     = p[k];
    fr.h[2 * v + 1] = p[k + 1];
  }
  return fr.v;
}

// B fragment (32x16 f16), LDS stored transposed as [n][k]: lane = N,
// lane group selects K half; 16 contiguous halves per lane.
DEV v16h frag_b_lds(const _Float16* base, int n0, int ldm, int lane) {
  const int n = lane & 15, g = lane >> 4;
  const _Float16* p = base + (size_t)(n0 + n) * ldm + g * 16;
  FragH fr;
#pragma unroll
  for (int v = 0; v < 16; ++v) fr.h[v] = p[v];
  return fr.v;
}

DEV v8f wmma_f16(v16h a, v16h b, v8f c) {
  return __builtin_amdgcn_wmma_f32_16x16x32_f16(false, a, false, b,
                                                (short)0, c, false, false);
}

// ---------------------------------------------------------------------------
// f32 -> f16 conversion, 4 elems/thread
// ---------------------------------------------------------------------------
__global__ __launch_bounds__(256)
void cvt_f32_f16(const float* __restrict__ in, _Float16* __restrict__ out, int n) {
  const int i = (blockIdx.x * 256 + threadIdx.x) * 4;
  if (i + 3 < n) {
    const float4 v = *(const float4*)(in + i);
    out[i]     = (_Float16)v.x;
    out[i + 1] = (_Float16)v.y;
    out[i + 2] = (_Float16)v.z;
    out[i + 3] = (_Float16)v.w;
  }
}

// ---------------------------------------------------------------------------
// Tiled f16 GEMM with WMMA, double-buffered async staging.
// BM=128, BN=64, BK=64; 256 threads = 8 waves; each wave: 2x2 16x16 tiles,
// 8 WMMAs per K-stage; one barrier per K-step.
// ---------------------------------------------------------------------------
constexpr int BM = 128, BN = 64, BK = 64;
constexpr int LDA_S = BK + 8; // 72 halves: rows stay 16B aligned
constexpr int LDB_S = BK + 8;

template <typename CT>
__global__ __launch_bounds__(256)
void gemm_f16_wmma(const _Float16* __restrict__ A, const _Float16* __restrict__ Bm,
                   CT* __restrict__ C, int Nc, int K) {
  __shared__ _Float16 As[2][BM * LDA_S];
  __shared__ _Float16 Bs[2][BN * LDB_S];

  const int tid  = threadIdx.x;
  const int lane = tid & 31, wid = tid >> 5;
  const int m0 = blockIdx.y * BM;
  const int n0 = blockIdx.x * BN;
  const int wm = (wid & 3) * 32;   // 4 waves span 128 rows
  const int wn = (wid >> 2) * 32;  // 2 waves span 64 cols

  v8f acc[2][2] = {};

  const int arow = tid >> 1, acol = (tid & 1) * 32; // 2 threads/row, 32 halves

  auto stage = [&](int k0, int buf) {
    const _Float16* ag = A + (size_t)(m0 + arow) * K + k0 + acol;
    _Float16* al = &As[buf][arow * LDA_S + acol];
#pragma unroll
    for (int i = 0; i < 4; ++i) async_copy16(ag + i * 8, al + i * 8);
    __builtin_prefetch(ag + BK, 0, 1); // global_prefetch_b8 for the next tile
#pragma unroll
    for (int i = 0; i < 16; ++i) {
      const int e = i * 256 + tid;
      const int n = e & 63, k = e >> 6;
      Bs[buf][n * LDB_S + k] = Bm[(size_t)(k0 + k) * Nc + n0 + n];
    }
  };

  int buf = 0;
  stage(0, 0);
  for (int k0 = 0; k0 < K; k0 += BK) {
    async_join();
    __syncthreads();
    if (k0 + BK < K) stage(k0 + BK, buf ^ 1);

#pragma unroll
    for (int kc = 0; kc < 2; ++kc) {
      const v16h a0 = frag_a_lds(As[buf] + kc * 32, wm,      LDA_S, lane);
      const v16h a1 = frag_a_lds(As[buf] + kc * 32, wm + 16, LDA_S, lane);
      const v16h b0 = frag_b_lds(Bs[buf] + kc * 32, wn,      LDB_S, lane);
      const v16h b1 = frag_b_lds(Bs[buf] + kc * 32, wn + 16, LDB_S, lane);
      acc[0][0] = wmma_f16(a0, b0, acc[0][0]);
      acc[0][1] = wmma_f16(a0, b1, acc[0][1]);
      acc[1][0] = wmma_f16(a1, b0, acc[1][0]);
      acc[1][1] = wmma_f16(a1, b1, acc[1][1]);
    }
    buf ^= 1;
  }

  const int cn = lane & 15, cg = lane >> 4;
#pragma unroll
  for (int i = 0; i < 2; ++i) {
#pragma unroll
    for (int j = 0; j < 2; ++j) {
      FragF fr; fr.v = acc[i][j];
      const int col = n0 + wn + j * 16 + cn;
#pragma unroll
      for (int r = 0; r < 8; ++r) {
        const int row = m0 + wm + i * 16 + r + 8 * cg;
        C[(size_t)row * Nc + col] = (CT)fr.f[r];
      }
    }
  }
}

// ---------------------------------------------------------------------------
// Split qkv -> head-major q/k/v, applying RoPE to k AND v (per reference).
// ---------------------------------------------------------------------------
__global__ __launch_bounds__(256)
void rope_split(const _Float16* __restrict__ qkv, const float* __restrict__ cosT,
                const float* __restrict__ sinT, _Float16* __restrict__ qh,
                _Float16* __restrict__ kh, _Float16* __restrict__ vh) {
  using namespace cfg;
  const int idx = blockIdx.x * 256 + threadIdx.x;  // B*T*NH*FR
  const int f = idx & (FR - 1);
  const int n = (idx >> 6) & (NH - 1);
  const int t = (idx >> 10) & (T - 1);
  const int b = idx >> 21;

  const size_t src = (size_t)(b * T + t) * TD;
  const int    off = n * HD + 2 * f;
  const size_t dst = ((size_t)(b * NH + n) * T + t) * HD + 2 * f;

  qh[dst]     = qkv[src + off];
  qh[dst + 1] = qkv[src + off + 1];

  const float c = cosT[t * FR + f];
  const float s = sinT[t * FR + f];

  const float kr = (float)qkv[src + D + off];
  const float ki = (float)qkv[src + D + off + 1];
  kh[dst]     = (_Float16)(kr * c - ki * s);
  kh[dst + 1] = (_Float16)(kr * s + ki * c);

  const float vr = (float)qkv[src + 2 * D + off];
  const float vi = (float)qkv[src + 2 * D + off + 1];
  vh[dst]     = (_Float16)(vr * c - vi * s);
  vh[dst + 1] = (_Float16)(vr * s + vi * c);
}

// ---------------------------------------------------------------------------
// Flash attention, double-buffered K/V staging, register-resident O and Q.
// One block per (head, 64-row Q tile), 256 threads = 8 waves.
// S tiles per wave: rows (wid>>2)*16 and (wid>>2+2)*16, col tile (wid&3)*16.
// PV strip per wave: rows (wid&3)*16, four col tiles (wid>>2)+2*i.
// ---------------------------------------------------------------------------
constexpr int QT = 64, KTile = 64;
constexpr int LQ = cfg::HD + 8;  // 136 halves
constexpr int LV = KTile + 8;    // 72 halves (V transposed [h][t])
constexpr int LSs = KTile + 4;   // 68 floats
constexpr int LP = KTile + 8;    // 72 halves
constexpr size_t FLASH_LDS =
    (size_t)64 * LQ * 2 +            // Qs
    (size_t)2 * 64 * LQ * 2 +        // Ks double buffer
    (size_t)2 * 128 * LV * 2 +       // Vt double buffer
    (size_t)64 * LSs * 4 +           // Ss
    (size_t)64 * LP * 2 +            // Ps
    (size_t)3 * 64 * 4;              // mi, li, alpha  => 116480 B

__global__ __launch_bounds__(256)
void flash_attn(const _Float16* __restrict__ qh, const _Float16* __restrict__ kh,
                const _Float16* __restrict__ vh, _Float16* __restrict__ oh) {
  using namespace cfg;
  extern __shared__ char smem[];
  _Float16* Qs  = (_Float16*)smem;             // 64 x LQ
  _Float16* Ks0 = Qs + 64 * LQ;                // 2 x (64 x LQ)
  _Float16* Vt0 = Ks0 + 2 * 64 * LQ;           // 2 x (128 x LV)
  float*    Ss  = (float*)(Vt0 + 2 * 128 * LV);// 64 x LSs
  _Float16* Ps  = (_Float16*)(Ss + 64 * LSs);  // 64 x LP
  float*    mi  = (float*)(Ps + 64 * LP);
  float*    li  = mi + 64;
  float*    al  = li + 64;

  const int tid = threadIdx.x, lane = tid & 31, wid = tid >> 5;
  const int bn = blockIdx.y, qt = blockIdx.x;
  const size_t head_base = (size_t)bn * T * HD;

  auto stage_kv = [&](int kt, int buf) {
    { // K tile, row-major [t'][h], async to LDS
      const int row = tid >> 2, c0 = (tid & 3) * 32;
      const _Float16* src = kh + head_base + (size_t)(kt * KTile + row) * HD + c0;
      _Float16* dst = Ks0 + buf * 64 * LQ + row * LQ + c0;
#pragma unroll
      for (int i = 0; i < 4; ++i) async_copy16(src + i * 8, dst + i * 8);
    }
    // V tile transposed into Vt[h][t'] (coalesced reads over h)
    _Float16* vt = Vt0 + buf * 128 * LV;
#pragma unroll 4
    for (int i = 0; i < 32; ++i) {
      const int e = i * 256 + tid;
      const int h = e & 127, tt = e >> 7;
      vt[h * LV + tt] = vh[head_base + (size_t)(kt * KTile + tt) * HD + h];
    }
  };

  { // Q tile (async) + stats init
    const int row = tid >> 2, c0 = (tid & 3) * 32;
    const _Float16* src = qh + head_base + (size_t)(qt * QT + row) * HD + c0;
    _Float16* dst = Qs + row * LQ + c0;
#pragma unroll
    for (int i = 0; i < 4; ++i) async_copy16(src + i * 8, dst + i * 8);
  }
  if (tid < 64) { mi[tid] = -3.0e38f; li[tid] = 0.f; }
  async_join();
  __syncthreads();

  // Hoist loop-invariant Q fragments into registers (2 row tiles x 4 K-chunks)
  v16h qa[2][4];
#pragma unroll
  for (int hv = 0; hv < 2; ++hv) {
    const int tm = ((wid >> 2) + 2 * hv) * 16;
#pragma unroll
    for (int kc = 0; kc < 4; ++kc)
      qa[hv][kc] = frag_a_lds(Qs + kc * 32, tm, LQ, lane);
  }

  stage_kv(0, 0);

  const float scale = 0.08838834764831845f; // 1/sqrt(128)
  const int tmi = wid & 3;                  // wave's fixed 16-row PV strip
  const int cg  = lane >> 4, cn = lane & 15;
  v8f accO[4] = {};

  int buf = 0;
  for (int kt = 0; kt <= qt; ++kt) {
    async_join();
    __syncthreads();
    if (kt < qt) stage_kv(kt + 1, buf ^ 1);
    const _Float16* KsCur = Ks0 + buf * 64 * LQ;
    const _Float16* VtCur = Vt0 + buf * 128 * LV;

    // ---- S = Q K^T : 16 tiles of 16x16, 2 per wave (A from registers) ----
    const int tn = tmi * 16;   // column tile = wave's PV row strip index
#pragma unroll
    for (int hv = 0; hv < 2; ++hv) {
      const int tm = ((wid >> 2) + 2 * hv) * 16;
      v8f acc = {};
#pragma unroll
      for (int kc = 0; kc < 4; ++kc)
        acc = wmma_f16(qa[hv][kc], frag_b_lds(KsCur + kc * 32, tn, LQ, lane), acc);
      FragF fr; fr.v = acc;
      const int col  = tn + cn;
      const int kcol = kt * KTile + col;
#pragma unroll
      for (int r = 0; r < 8; ++r) {
        const int row  = tm + r + 8 * cg;
        const int qrow = qt * QT + row;
        Ss[row * LSs + col] = (kcol <= qrow) ? fr.f[r] * scale : -3.0e38f;
      }
    }
    __syncthreads();

    // ---- online softmax: one thread per row; publish alpha ----
    if (tid < 64) {
      const int row = tid;
      float mnew = mi[row];
      for (int j = 0; j < KTile; ++j) mnew = fmaxf(mnew, Ss[row * LSs + j]);
      const float alpha = __expf(mi[row] - mnew);
      float lsum = 0.f;
      for (int j = 0; j < KTile; ++j) {
        const float p = __expf(Ss[row * LSs + j] - mnew);
        Ps[row * LP + j] = (_Float16)p;
        lsum += p;
      }
      li[row] = li[row] * alpha + lsum;
      mi[row] = mnew;
      al[row] = alpha;
    }
    __syncthreads();

    // ---- O = O*alpha + P V, all in registers ----
    float alpha_r[8];
#pragma unroll
    for (int r = 0; r < 8; ++r) alpha_r[r] = al[tmi * 16 + r + 8 * cg];

    const v16h pa0 = frag_a_lds(Ps,      tmi * 16, LP, lane);
    const v16h pa1 = frag_a_lds(Ps + 32, tmi * 16, LP, lane);
#pragma unroll
    for (int i4 = 0; i4 < 4; ++i4) {
      const int tni = (wid >> 2) + 2 * i4;  // wave's 4 column tiles
      FragF fr; fr.v = accO[i4];
#pragma unroll
      for (int r = 0; r < 8; ++r) fr.f[r] *= alpha_r[r];
      v8f acc = fr.v;
      acc = wmma_f16(pa0, frag_b_lds(VtCur,      tni * 16, LV, lane), acc);
      acc = wmma_f16(pa1, frag_b_lds(VtCur + 32, tni * 16, LV, lane), acc);
      accO[i4] = acc;
    }
    buf ^= 1;
    // no barrier needed: next iteration's loop-top barrier separates
  }

  // ---- epilogue: normalize by l, store o in [B,T,N*H] layout ----
  const int b = bn / NH, n = bn % NH;
  float lival[8];
#pragma unroll
  for (int r = 0; r < 8; ++r) lival[r] = li[tmi * 16 + r + 8 * cg];
#pragma unroll
  for (int i4 = 0; i4 < 4; ++i4) {
    const int tni = (wid >> 2) + 2 * i4;
    FragF fr; fr.v = accO[i4];
    const int col = tni * 16 + cn;
#pragma unroll
    for (int r = 0; r < 8; ++r) {
      const int row = tmi * 16 + r + 8 * cg;
      const int tg  = qt * QT + row;
      oh[((size_t)(b * T + tg)) * D + n * HD + col] = (_Float16)(fr.f[r] / lival[r]);
    }
  }
}

// ---------------------------------------------------------------------------
extern "C" void kernel_launch(void* const* d_in, const int* in_sizes, int n_in,
                              void* d_out, int out_size, void* d_ws, size_t ws_size,
                              hipStream_t stream) {
  using namespace cfg;
  (void)in_sizes; (void)n_in; (void)out_size; (void)ws_size;
  const float* x      = (const float*)d_in[0];
  // d_in[1] is the causal mask; causality is computed analytically.
  const float* cosT   = (const float*)d_in[2];
  const float* sinT   = (const float*)d_in[3];
  const float* w_attn = (const float*)d_in[4];
  const float* w_out  = (const float*)d_in[5];

  char* ws = (char*)d_ws;
  size_t off = 0;
  auto carve = [&](size_t bytes) -> void* {
    void* p = ws + off;
    off = (off + bytes + 255) & ~(size_t)255;
    return p;
  };
  _Float16* xh   = (_Float16*)carve((size_t)B * T * D * 2);
  _Float16* wah  = (_Float16*)carve((size_t)D * TD * 2);
  _Float16* woh  = (_Float16*)carve((size_t)D * D * 2);
  _Float16* qkvh = (_Float16*)carve((size_t)B * T * TD * 2);
  _Float16* qh   = (_Float16*)carve((size_t)B * T * D * 2);
  _Float16* kh   = (_Float16*)carve((size_t)B * T * D * 2);
  _Float16* vh   = (_Float16*)carve((size_t)B * T * D * 2);
  _Float16* oh   = (_Float16*)carve((size_t)B * T * D * 2);

  const int n_x = B * T * D, n_wa = D * TD, n_wo = D * D;
  cvt_f32_f16<<<dim3(n_x  / 1024), 256, 0, stream>>>(x,      xh,  n_x);
  cvt_f32_f16<<<dim3(n_wa / 1024), 256, 0, stream>>>(w_attn, wah, n_wa);
  cvt_f32_f16<<<dim3(n_wo / 1024), 256, 0, stream>>>(w_out,  woh, n_wo);

  // qkv = x @ w_attn  (M=4096, N=6144, K=2048)
  gemm_f16_wmma<_Float16><<<dim3(TD / BN, (B * T) / BM), 256, 0, stream>>>(
      xh, wah, qkvh, TD, D);

  // split + RoPE(k, v)
  rope_split<<<dim3((B * T * NH * FR) / 256), 256, 0, stream>>>(
      qkvh, cosT, sinT, qh, kh, vh);

  // flash attention per (head, 64-row q tile)
  flash_attn<<<dim3(T / QT, B * NH), 256, FLASH_LDS, stream>>>(qh, kh, vh, oh);

  // out = o @ w_out  (M=4096, N=2048, K=2048), f32 output
  gemm_f16_wmma<float><<<dim3(D / BN, (B * T) / BM), 256, 0, stream>>>(
      oh, woh, (float*)d_out, D, D);
}
